// NCA_38869454029120
// MI455X (gfx1250) — compile-verified, matrix operand order checked by reading, not run on hardware
//
#include <hip/hip_runtime.h>

typedef __attribute__((ext_vector_type(16))) _Float16 v16h;
typedef __attribute__((ext_vector_type(8)))  _Float16 v8h;
typedef __attribute__((ext_vector_type(8)))  float    v8f;
typedef __attribute__((ext_vector_type(4)))  unsigned int u32x4;
typedef __attribute__((ext_vector_type(4)))  int          i32x4;
typedef __attribute__((ext_vector_type(8)))  int          i32x8;

#define BATCH 1024
#define IMG   28
#define PIX   (IMG*IMG)            // 784
#define CH    16
#define HID   128
#define TILES_PER_IMG (PIX/16)     // 49
#define NTILES (BATCH*TILES_PER_IMG) // 50176
#define WAVES_PER_BLK 8
#define BLKSZ 256
#define NBLK  (NTILES/WAVES_PER_BLK) // 6272
#define LN_N  ((float)(PIX*HID))   // 100352
#define STEPS 10
#define ALPHA 0.2f
#define EPS   1e-3f

#define WPACK_HALVES (5*8*32*16)   // 20480 f16 = 40960 B
#define WFC1_HALVES  (4*32*16)     // 2048 f16 = 4096 B

// workspace byte offsets
#define STATS_OFF 0u               // [10][1024][2] f32 = 81920 B
#define BCOMB_OFF 81920u           // [128] f32
#define WPACK_OFF 82432u           // 40960 B (16B aligned)
#define WFC1_OFF  123392u          // 4096 B
#define DELTA_OFF 131072u          // [1024*784*16] f32 = 51,380,224 B

#ifndef __has_builtin
#define __has_builtin(x) 0
#endif
#if __has_builtin(__builtin_amdgcn_tensor_load_to_lds) && \
    __has_builtin(__builtin_amdgcn_s_wait_tensorcnt)
#define HAVE_TDM 1
#else
#define HAVE_TDM 0
#endif

// K index inside a 16x32 f16 A/B fragment for (lane, vector element e):
// lane<16: e0..7 -> K 0..7,  e8..15 -> K 16..23
// lane>=16: e0..7 -> K 8..15, e8..15 -> K 24..31
__device__ inline int kmap(int lane, int e) {
    return e + ((e >> 3) << 3) + ((lane >> 4) << 3);
}

#if HAVE_TDM
// ---------------------------------------------------------------------------
// Tensor Data Mover: DMA a contiguous block (multiple of 8B) global -> LDS as
// a 1-D tile.  D# per CDNA5 ISA.  Not in a cluster, so
// workgroup_mask = 0.  LDS offset = low 32 bits of the flat shared pointer
// (ISA §10.2: LDS aperture maps addr[31:0] to the wave's LDS offset).
// ---------------------------------------------------------------------------
__device__ inline void tdm_load_to_lds(const void* gsrc, void* ldst,
                                       unsigned elems8B)
{
    unsigned lds_off = (unsigned)(unsigned long long)(uintptr_t)ldst;
    unsigned long long ga = (unsigned long long)(uintptr_t)gsrc;
    u32x4 g0;
    g0[0] = 1u;                                        // count=1
    g0[1] = lds_off;                                   // lds_addr (bytes)
    g0[2] = (unsigned)(ga & 0xFFFFFFFFu);              // global_addr[31:0]
    g0[3] = (unsigned)((ga >> 32) & 0x01FFFFFFu) | (2u << 30); // addr[56:32], type=2
    i32x8 g1;
    g1[0] = (int)(3u << 16);                           // mask=0, data_size=3 (8B)
    g1[1] = (int)((elems8B & 0xFFFFu) << 16);          // tensor_dim0[15:0] @bits63:48
    g1[2] = (int)(((elems8B >> 16) & 0xFFFFu)          // tensor_dim0[31:16]
                  | (1u << 16));                       // tensor_dim1 = 1
    g1[3] = (int)((elems8B & 0xFFFFu) << 16);          // tile_dim0 @bits127:112
    g1[4] = 1;                                         // tile_dim1 = 1
    g1[5] = (int)elems8B;                              // tensor_dim0_stride[31:0]
    g1[6] = 0;
    g1[7] = 0;
    i32x4 zz = {0, 0, 0, 0};
#if __clang_major__ >= 23
    i32x8 z8 = {0, 0, 0, 0, 0, 0, 0, 0};
    __builtin_amdgcn_tensor_load_to_lds(g0, g1, zz, zz, z8, 0);
#else
    __builtin_amdgcn_tensor_load_to_lds(g0, g1, zz, zz, 0);
#endif
}
#endif

// Stage a packed-f16 weight block into LDS (TDM if available, else coop copy).
__device__ inline void stage_block(const _Float16* __restrict__ g,
                                   _Float16* l, int halves, int tid)
{
#if HAVE_TDM
    if (tid < 32) tdm_load_to_lds(g, l, (unsigned)(halves / 4));
#else
    const uint4* src = (const uint4*)g;
    uint4*       dst = (uint4*)l;
    for (int i = tid; i < halves / 8; i += BLKSZ) dst[i] = src[i];
#endif
}

__device__ inline void stage_fence(int tid)
{
#if HAVE_TDM
    if (tid < 32) __builtin_amdgcn_s_wait_tensorcnt(0);
#endif
    __syncthreads();
}

// ---------------------------------------------------------------------------
// One-time weight folding + packing into wave32 WMMA fragment layout.
// Wcomb[tap][kc][d] = sum_j wp0[tap][kc][j]*wfc0[16+j][d]
//                   + sum_j wp1[tap][kc][j]*wfc0[32+j][d]
//                   + (tap==4)*wfc0[kc][d]           (taps 0..8; tap 9 = zero pad)
// ---------------------------------------------------------------------------
__global__ void pack_weights_kernel(const float* __restrict__ wp0,
                                    const float* __restrict__ bp0,
                                    const float* __restrict__ wp1,
                                    const float* __restrict__ bp1,
                                    const float* __restrict__ wfc0,
                                    const float* __restrict__ bfc0,
                                    const float* __restrict__ wfc1,
                                    _Float16* __restrict__ wpack,
                                    float* __restrict__ bcomb,
                                    _Float16* __restrict__ wfc1p)
{
    int tid = blockIdx.x * blockDim.x + threadIdx.x;
    if (tid < WPACK_HALVES) {                // conv B fragments: [kb][nb][lane][e]
        int e = tid & 15, lane = (tid >> 4) & 31, nb = (tid >> 9) & 7, kb = tid >> 12;
        int gk  = kb * 32 + kmap(lane, e);   // 0..159
        int tap = gk >> 4, kc = gk & 15;
        int d   = nb * 16 + (lane & 15);
        float v = 0.f;
        if (tap < 9) {
            const float* p0 = wp0 + (tap * 16 + kc) * 16;
            const float* p1 = wp1 + (tap * 16 + kc) * 16;
            #pragma unroll
            for (int j = 0; j < 16; ++j) {
                v += p0[j] * wfc0[(16 + j) * HID + d];
                v += p1[j] * wfc0[(32 + j) * HID + d];
            }
            if (tap == 4) v += wfc0[kc * HID + d];
        }
        wpack[tid] = (_Float16)v;
    } else if (tid < WPACK_HALVES + 128) {   // folded bias, 128 entries
        int d = tid - WPACK_HALVES;
        float v = bfc0[d];
        #pragma unroll
        for (int j = 0; j < 16; ++j) {
            v += bp0[j] * wfc0[(16 + j) * HID + d];
            v += bp1[j] * wfc0[(32 + j) * HID + d];
        }
        bcomb[d] = v;
    } else if (tid < WPACK_HALVES + 128 + WFC1_HALVES) { // fc1 B fragments
        int j = tid - (WPACK_HALVES + 128);
        int e = j & 15, lane = (j >> 4) & 31, kb = j >> 9;
        int d = kb * 32 + kmap(lane, e);     // 0..127
        int c = lane & 15;
        wfc1p[j] = (_Float16)wfc1[d * 16 + c];
    }
}

// seed(): pad 3 input channels to 16
__global__ void seed_kernel(const float* __restrict__ x, float* __restrict__ z)
{
    int tid = blockIdx.x * blockDim.x + threadIdx.x;  // 1024*784*16 total
    int c = tid & 15;
    int p = tid >> 4;                                  // b*784 + pix
    z[tid] = (c < 3) ? x[p * 3 + c] : 0.f;
}

// ---------------------------------------------------------------------------
// Core: one wave computes h = leakyrelu(conv16->128(z) + b) for a 16-pixel tile
// as 8 WMMA D-fragments (acc[nb]: lane = out-channel within 16, vgpr = pixel).
// B fragments come from LDS (staged once per block).
// ---------------------------------------------------------------------------
__device__ inline void compute_h_tile(const float* __restrict__ z,
                                      const _Float16* wB,   // LDS
                                      const float* __restrict__ bcomb,
                                      int b, int basePix, int lane, v8f acc[8])
{
    const int m     = lane & 15;
    const int cbase = (lane >> 4) << 3;     // lanes 16-31 carry channels 8..15
    const int pix   = basePix + m;
    const int py    = pix / IMG, px = pix % IMG;

    v16h A[5];                               // 5 K-blocks of 32 (taps packed 2/blk)
    #pragma unroll
    for (int kb = 0; kb < 5; ++kb) {
        #pragma unroll
        for (int half = 0; half < 2; ++half) {
            const int tap = kb * 2 + half;   // 0..9 (9 = zero pad)
            float4 f0 = {0.f, 0.f, 0.f, 0.f};
            float4 f1 = {0.f, 0.f, 0.f, 0.f};
            if (tap < 9) {
                const int ny = py + tap / 3 - 1;
                const int nx = px + tap % 3 - 1;
                if (ny >= 0 && ny < IMG && nx >= 0 && nx < IMG) {
                    const float4* p = (const float4*)(z +
                        ((size_t)b * PIX + ny * IMG + nx) * CH + cbase);
                    f0 = p[0]; f1 = p[1];
                }
            }
            const int eb = half * 8;
            float t[8] = {f0.x, f0.y, f0.z, f0.w, f1.x, f1.y, f1.z, f1.w};
            #pragma unroll
            for (int i = 0; i < 8; ++i) A[kb][eb + i] = (_Float16)t[i];
        }
    }

    #pragma unroll
    for (int nb = 0; nb < 8; ++nb) {
        const float bias = bcomb[nb * 16 + m];
        v8f c = {bias, bias, bias, bias, bias, bias, bias, bias};
        #pragma unroll
        for (int kb = 0; kb < 5; ++kb) {
            v16h Bf = *(const v16h*)(wB + ((kb * 8 + nb) * 32 + lane) * 16);
            c = __builtin_amdgcn_wmma_f32_16x16x32_f16(
                    false, A[kb], false, Bf, (short)0, c, false, false);
        }
        #pragma unroll
        for (int i = 0; i < 8; ++i) {
            float v = c[i];
            c[i] = v >= 0.f ? v : ALPHA * v;
        }
        acc[nb] = c;
    }
}

// Pass A: conv + leakyrelu, accumulate per-sample LayerNorm statistics.
__global__ void __launch_bounds__(BLKSZ)
conv_stats_kernel(const float* __restrict__ z,
                  const _Float16* __restrict__ wpack,
                  const float* __restrict__ bcomb,
                  float* __restrict__ stats)       // [1024][2] for this step
{
    __shared__ _Float16 wLDS[WPACK_HALVES];          // 40 KB
    stage_block(wpack, wLDS, WPACK_HALVES, threadIdx.x);
    stage_fence(threadIdx.x);

    const int wave    = blockIdx.x * WAVES_PER_BLK + (threadIdx.x >> 5);
    const int lane    = threadIdx.x & 31;
    const int b       = wave / TILES_PER_IMG;
    const int basePix = (wave % TILES_PER_IMG) * 16;

    v8f acc[8];
    compute_h_tile(z, wLDS, bcomb, b, basePix, lane, acc);

    float s = 0.f, q = 0.f;
    #pragma unroll
    for (int nb = 0; nb < 8; ++nb)
        #pragma unroll
        for (int i = 0; i < 8; ++i) {
            float v = acc[nb][i];
            s += v; q += v * v;
        }
    #pragma unroll
    for (int off = 16; off > 0; off >>= 1) {
        s += __shfl_xor(s, off, 32);
        q += __shfl_xor(q, off, 32);
    }
    if (lane == 0) {
        atomicAdd(&stats[b * 2 + 0], s);
        atomicAdd(&stats[b * 2 + 1], q);
    }
}

// Pass B: recompute h, normalize, fc1 via WMMA, write masked delta.
__global__ void __launch_bounds__(BLKSZ)
nca_update_kernel(const float* __restrict__ z,
                  const _Float16* __restrict__ wpack,
                  const float* __restrict__ bcomb,
                  const float* __restrict__ stats,
                  const float* __restrict__ gamma,
                  const float* __restrict__ beta,
                  const _Float16* __restrict__ wfc1p,
                  float* __restrict__ delta,
                  unsigned step_seed)
{
    __shared__ _Float16 wLDS[WPACK_HALVES];             // 40 KB
    __shared__ _Float16 w1LDS[WFC1_HALVES];             // 4 KB
    __shared__ _Float16 hT[WAVES_PER_BLK * 16 * HID];   // 32 KB
    stage_block(wpack, wLDS, WPACK_HALVES, threadIdx.x);
    stage_block(wfc1p, w1LDS, WFC1_HALVES, threadIdx.x);
    stage_fence(threadIdx.x);

    const int wslot   = threadIdx.x >> 5;
    const int wave    = blockIdx.x * WAVES_PER_BLK + wslot;
    const int lane    = threadIdx.x & 31;
    const int b       = wave / TILES_PER_IMG;
    const int basePix = (wave % TILES_PER_IMG) * 16;

    v8f acc[8];
    compute_h_tile(z, wLDS, bcomb, b, basePix, lane, acc);

    const float sum  = stats[b * 2 + 0];
    const float sq   = stats[b * 2 + 1];
    const float mean = sum * (1.f / LN_N);
    const float var  = sq * (1.f / LN_N) - mean * mean;
    const float rstd = rsqrtf(var + EPS);

    // Normalize + gamma/beta, transpose D-layout -> A-layout through LDS.
    _Float16* ht = hT + wslot * 16 * HID;
    const int n      = lane & 15;
    const int mShift = (lane >> 4) << 3;
    #pragma unroll
    for (int nb = 0; nb < 8; ++nb) {
        const int d = nb * 16 + n;
        #pragma unroll
        for (int v = 0; v < 8; ++v) {
            const int m   = v + mShift;
            const int pix = basePix + m;
            const float g  = gamma[pix * HID + d];
            const float be = beta[pix * HID + d];
            ht[m * HID + d] = (_Float16)((acc[nb][v] - mean) * rstd * g + be);
        }
    }
    __syncthreads();

    // dx[16x16] = hnorm[16x128] @ wfc1[128x16], 4 chained K=32 WMMAs
    v8f dx = {0.f, 0.f, 0.f, 0.f, 0.f, 0.f, 0.f, 0.f};
    const int am   = lane & 15;
    const int kadd = (lane >> 4) << 3;
    #pragma unroll
    for (int kb = 0; kb < 4; ++kb) {
        const int d0 = kb * 32 + kadd;
        v8h lo = *(const v8h*)(ht + am * HID + d0);
        v8h hi = *(const v8h*)(ht + am * HID + d0 + 16);
        v16h a;
        #pragma unroll
        for (int i = 0; i < 8; ++i) { a[i] = lo[i]; a[i + 8] = hi[i]; }
        v16h Bf = *(const v16h*)(w1LDS + (kb * 32 + lane) * 16);
        dx = __builtin_amdgcn_wmma_f32_16x16x32_f16(
                 false, a, false, Bf, (short)0, dx, false, false);
    }

    // stochastic fire mask (per pixel), store masked delta
    #pragma unroll
    for (int v = 0; v < 8; ++v) {
        const int m   = v + mShift;
        const int pix = basePix + m;
        unsigned sd = (unsigned)(b * PIX + pix) + step_seed;
        sd = sd * 747796405u + 2891336453u;
        unsigned w = ((sd >> ((sd >> 28) + 4u)) ^ sd) * 277803737u;
        w = (w >> 22) ^ w;
        const float mask = (w >> 31) ? 1.f : 0.f;
        delta[((size_t)b * PIX + pix) * CH + n] = dx[v] * mask;
    }
}

// Pass C: residual update (separate kernel so conv halo reads never race writes)
__global__ void apply_delta_kernel(float* __restrict__ z,
                                   const float* __restrict__ delta)
{
    size_t i = (size_t)blockIdx.x * blockDim.x + threadIdx.x;
    z[i] += delta[i];
}

extern "C" void kernel_launch(void* const* d_in, const int* in_sizes, int n_in,
                              void* d_out, int out_size, void* d_ws, size_t ws_size,
                              hipStream_t stream)
{
    const float* x     = (const float*)d_in[0];
    const float* wp0   = (const float*)d_in[1];
    const float* bp0   = (const float*)d_in[2];
    const float* wp1   = (const float*)d_in[3];
    const float* bp1   = (const float*)d_in[4];
    const float* wfc0  = (const float*)d_in[5];
    const float* bfc0  = (const float*)d_in[6];
    const float* wfc1  = (const float*)d_in[7];
    const float* gamma = (const float*)d_in[8];
    const float* beta  = (const float*)d_in[9];
    // d_in[10] = steps (known constant 10)

    char*     ws    = (char*)d_ws;
    float*    stats = (float*)(ws + STATS_OFF);
    float*    bcomb = (float*)(ws + BCOMB_OFF);
    _Float16* wpack = (_Float16*)(ws + WPACK_OFF);
    _Float16* wfc1p = (_Float16*)(ws + WFC1_OFF);
    float*    delta = (float*)(ws + DELTA_OFF);
    float*    z     = (float*)d_out;              // z lives in d_out

    hipMemsetAsync(stats, 0, STEPS * BATCH * 2 * sizeof(float), stream);
    seed_kernel<<<(BATCH * PIX * CH) / 256, 256, 0, stream>>>(x, z);
    pack_weights_kernel<<<89, 256, 0, stream>>>(wp0, bp0, wp1, bp1, wfc0, bfc0,
                                                wfc1, wpack, bcomb, wfc1p);

    for (int s = 0; s < STEPS; ++s) {
        float* st = stats + s * BATCH * 2;
        conv_stats_kernel<<<NBLK, BLKSZ, 0, stream>>>(z, wpack, bcomb, st);
        nca_update_kernel<<<NBLK, BLKSZ, 0, stream>>>(
            z, wpack, bcomb, st, gamma, beta, wfc1p, delta,
            0x9E3779B9u * (unsigned)(s + 1));
        apply_delta_kernel<<<(BATCH * PIX * CH) / 256, 256, 0, stream>>>(z, delta);
    }
}
